// DGIGCN_8211977470552
// MI455X (gfx1250) — matrix operand order repaired and checked by reading, compiled, not run on hardware
//
#include <hip/hip_runtime.h>
#include <hip/hip_bf16.h>
#include <stddef.h>

// ---------------------------------------------------------------------------
// DGI-GCN layer for MI455X (gfx1250):
//   0) one-shot split of X and W into (hi, lo) bf16 planes in workspace
//   1) x_fts = X @ W^T    -> bf16-split WMMA GEMM, conversion-free inner loop
//   2) out = spmm(adj, x_fts) + bias, PReLU  -> CSR segment-sum, no atomics
// ---------------------------------------------------------------------------

typedef __attribute__((ext_vector_type(16))) __bf16 v16bf;
typedef __attribute__((ext_vector_type(8)))  __bf16 v8bf;
typedef __attribute__((ext_vector_type(4)))  __bf16 v4bf;
typedef __attribute__((ext_vector_type(8)))  float  v8f;

union frag16 { v16bf v; v8bf h[2]; };

#define IN_CH 512
#define HID   512

// ---------------------------------------------------------------------------
// Elementwise split: src f32 -> hi = bf16(x), lo = bf16(x - hi). 4 elems/thread.
// ---------------------------------------------------------------------------
__global__ void __launch_bounds__(256)
split_f32_to_bf16(const float* __restrict__ src, __bf16* __restrict__ hi,
                  __bf16* __restrict__ lo, int n4) {
    int i = blockIdx.x * blockDim.x + threadIdx.x;
    if (i >= n4) return;
    const float4 s = *(const float4*)(src + (size_t)i * 4);
    v4bf h, l;
    float f[4] = {s.x, s.y, s.z, s.w};
    #pragma unroll
    for (int e = 0; e < 4; ++e) {
        __bf16 hb = (__bf16)f[e];
        h[e] = hb;
        l[e] = (__bf16)(f[e] - (float)hb);
    }
    *(v4bf*)(hi + (size_t)i * 4) = h;
    *(v4bf*)(lo + (size_t)i * 4) = l;
}

// ---------------------------------------------------------------------------
// GEMM: Y[m][n] = sum_k X[m][k] * W[n][k], operands pre-split to bf16 hi/lo.
// One wave -> 16x64 output strip (4 accumulators), K stepped by 32.
// Inner loop: 20x global_load_b128 + 12x v_wmma_f32_16x16x32_bf16, no cvt.
// ---------------------------------------------------------------------------
__global__ void __launch_bounds__(256)
gemm_xwT_wmma(const __bf16* __restrict__ Xhi, const __bf16* __restrict__ Xlo,
              const __bf16* __restrict__ Whi, const __bf16* __restrict__ Wlo,
              float* __restrict__ Y, int M, int K, int N, int mtiles, int nsuper) {
    const int lane  = threadIdx.x & 31;
    const int wave  = blockIdx.x * (blockDim.x >> 5) + (threadIdx.x >> 5);
    const int mtile = wave / nsuper;
    const int nsup  = wave % nsuper;
    if (mtile >= mtiles) return;     // uniform per wave; EXEC stays all-ones

    const int l15   = lane & 15;
    const int hi16  = lane >> 4;                 // 0 or 1
    const int mrow  = mtile * 16 + l15;          // A layout: M = lane & 15
    const int koffA = hi16 * 8;                  // A: lanes 16-31 hold K+8 / K+24
    const int koffB = hi16 * 8;                  // B: lanes 16-31 hold K+8 / K+24

    const __bf16* xh = Xhi + (size_t)mrow * K + koffA;
    const __bf16* xl = Xlo + (size_t)mrow * K + koffA;

    v8f acc[4];
    #pragma unroll
    for (int a = 0; a < 4; ++a) acc[a] = (v8f){0.f,0.f,0.f,0.f,0.f,0.f,0.f,0.f};

    for (int k0 = 0; k0 < K; k0 += 32) {
        // ---- A fragments: elements 0..7 <- K=k0+koffA.., 8..15 <- K=k0+16+koffA..
        frag16 a_hi, a_lo;
        a_hi.h[0] = *(const v8bf*)(xh + k0);
        a_hi.h[1] = *(const v8bf*)(xh + k0 + 16);
        a_lo.h[0] = *(const v8bf*)(xl + k0);
        a_lo.h[1] = *(const v8bf*)(xl + k0 + 16);

        // ---- 4 B fragments / accumulators reuse the A fragments
        #pragma unroll
        for (int a = 0; a < 4; ++a) {
            const int ncol = nsup * 64 + a * 16 + l15;       // B col n == W row n
            const __bf16* wh = Whi + (size_t)ncol * K + k0 + koffB;
            const __bf16* wl = Wlo + (size_t)ncol * K + k0 + koffB;
            frag16 b_hi, b_lo;
            b_hi.h[0] = *(const v8bf*)(wh);
            b_hi.h[1] = *(const v8bf*)(wh + 16);
            b_lo.h[0] = *(const v8bf*)(wl);
            b_lo.h[1] = *(const v8bf*)(wl + 16);

            acc[a] = __builtin_amdgcn_wmma_f32_16x16x32_bf16(
                         false, a_hi.v, false, b_hi.v, (short)0, acc[a], false, false);
            acc[a] = __builtin_amdgcn_wmma_f32_16x16x32_bf16(
                         false, a_hi.v, false, b_lo.v, (short)0, acc[a], false, false);
            acc[a] = __builtin_amdgcn_wmma_f32_16x16x32_bf16(
                         false, a_lo.v, false, b_hi.v, (short)0, acc[a], false, false);
        }
    }

    // ---- store: C/D layout -> row = vgpr + 8*hi16, col = lane & 15
    const int rbase = mtile * 16 + hi16 * 8;
    #pragma unroll
    for (int a = 0; a < 4; ++a) {
        const int col = nsup * 64 + a * 16 + l15;
        #pragma unroll
        for (int i = 0; i < 8; ++i)
            Y[(size_t)(rbase + i) * N + col] = acc[a][i];
    }
}

// ---------------------------------------------------------------------------
// row_ptr[n] = lower_bound(adj_row, n)  (adj_row is sorted)
// ---------------------------------------------------------------------------
__global__ void build_row_ptr(const int* __restrict__ rows, int nEdges, int nNodes,
                              int* __restrict__ row_ptr) {
    int n = blockIdx.x * blockDim.x + threadIdx.x;
    if (n > nNodes) return;
    int lo = 0, hi = nEdges;
    while (lo < hi) {
        int mid = (lo + hi) >> 1;
        if (rows[mid] < n) lo = mid + 1; else hi = mid;
    }
    row_ptr[n] = lo;
}

// ---------------------------------------------------------------------------
// out[n][ch..ch+3] = prelu( sum_e val[e]*x_fts[col[e]][ch..ch+3] + bias )
// Thread = (node, 4-channel chunk): broadcast val/col, coalesced row gathers.
// ---------------------------------------------------------------------------
__global__ void __launch_bounds__(256)
spmm_bias_prelu(const float* __restrict__ xf, const float* __restrict__ vals,
                const int* __restrict__ cols, const int* __restrict__ row_ptr,
                const float* __restrict__ bias, const float* __restrict__ prelu_a,
                float* __restrict__ out, int nNodes, int H) {
    const int chunks = H >> 2;                      // 128
    int t = blockIdx.x * blockDim.x + threadIdx.x;
    int n = t / chunks;
    int c = t % chunks;
    if (n >= nNodes) return;
    const int ch = c << 2;

    float4 acc = make_float4(0.f, 0.f, 0.f, 0.f);
    const int e0 = row_ptr[n];
    const int e1 = row_ptr[n + 1];
    for (int e = e0; e < e1; ++e) {
        const float v = vals[e];
        const int   k = cols[e];
        const float4 g = *(const float4*)(xf + (size_t)k * H + ch);
        acc.x += v * g.x; acc.y += v * g.y; acc.z += v * g.z; acc.w += v * g.w;
    }
    const float4 b = *(const float4*)(bias + ch);
    const float slope = prelu_a[0];
    acc.x += b.x; acc.y += b.y; acc.z += b.z; acc.w += b.w;
    acc.x = acc.x >= 0.f ? acc.x : slope * acc.x;
    acc.y = acc.y >= 0.f ? acc.y : slope * acc.y;
    acc.z = acc.z >= 0.f ? acc.z : slope * acc.z;
    acc.w = acc.w >= 0.f ? acc.w : slope * acc.w;
    *(float4*)(out + (size_t)n * H + ch) = acc;
}

// ---------------------------------------------------------------------------
extern "C" void kernel_launch(void* const* d_in, const int* in_sizes, int n_in,
                              void* d_out, int out_size, void* d_ws, size_t ws_size,
                              hipStream_t stream) {
    const float* x        = (const float*)d_in[0];   // [N_NODES, IN_CH]
    const float* fc_w     = (const float*)d_in[1];   // [HID, IN_CH]
    const float* bias     = (const float*)d_in[2];   // [HID]
    const float* prelu_a  = (const float*)d_in[3];   // [1]
    const float* adj_vals = (const float*)d_in[4];   // [E]
    const int*   adj_row  = (const int*)d_in[5];     // [E], sorted
    const int*   adj_col  = (const int*)d_in[6];     // [E]
    float*       out      = (float*)d_out;           // [N_NODES, HID]

    const int nNodes = in_sizes[0] / IN_CH;          // 10000
    const int nEdges = in_sizes[4];                  // 160000
    const size_t nX  = (size_t)nNodes * IN_CH;       // 5.12M
    const size_t nW  = (size_t)HID * IN_CH;          // 262144

    // Workspace layout (256B-aligned sections):
    //   [x_fts f32 N*H][Xhi bf16][Xlo bf16][Whi bf16][Wlo bf16][row_ptr int N+1]
    char* ws = (char*)d_ws;
    size_t off = 0;
    auto take = [&](size_t bytes) { char* p = ws + off;
                                    off = (off + bytes + 255) & ~(size_t)255; return p; };
    float*  xf  = (float*)take(nX * sizeof(float));
    __bf16* Xhi = (__bf16*)take(nX * sizeof(__bf16));
    __bf16* Xlo = (__bf16*)take(nX * sizeof(__bf16));
    __bf16* Whi = (__bf16*)take(nW * sizeof(__bf16));
    __bf16* Wlo = (__bf16*)take(nW * sizeof(__bf16));
    int* row_ptr = (int*)take((size_t)(nNodes + 1) * sizeof(int));
    (void)ws_size;

    // 0) one-shot fp32 -> (hi, lo) bf16 splits
    {
        const int x4 = (int)(nX / 4);
        split_f32_to_bf16<<<(x4 + 255) / 256, 256, 0, stream>>>(x, Xhi, Xlo, x4);
        const int w4 = (int)(nW / 4);
        split_f32_to_bf16<<<(w4 + 255) / 256, 256, 0, stream>>>(fc_w, Whi, Wlo, w4);
    }
    // 1) GEMM x_fts = X @ W^T  (one wave per 16x64 strip)
    {
        const int mtiles = (nNodes + 15) / 16;       // 625
        const int nsuper = HID / 64;                 // 8
        const int waves  = mtiles * nsuper;          // 5000
        const int blocks = (waves + 7) / 8;          // 256 thr = 8 waves/block
        gemm_xwT_wmma<<<blocks, 256, 0, stream>>>(Xhi, Xlo, Whi, Wlo, xf,
                                                  nNodes, IN_CH, HID, mtiles, nsuper);
    }
    // 2) row_ptr via binary search over sorted adj_row
    build_row_ptr<<<(nNodes + 1 + 255) / 256, 256, 0, stream>>>(adj_row, nEdges,
                                                                nNodes, row_ptr);
    // 3) segment-sum gather + bias + PReLU, fused
    {
        const long threads = (long)nNodes * (HID / 4);   // 1.28M
        const int  blocks  = (int)((threads + 255) / 256);
        spmm_bias_prelu<<<blocks, 256, 0, stream>>>(xf, adj_vals, adj_col, row_ptr,
                                                    bias, prelu_a, out, nNodes, HID);
    }
}
